// ComplexityAttention_4252017623495
// MI455X (gfx1250) — matrix operand order repaired and checked by reading, not compiled, not run on hardware
//
#include <hip/hip_runtime.h>

typedef __bf16 bf16;
typedef __attribute__((ext_vector_type(16))) __bf16 v16bf;
typedef __attribute__((ext_vector_type(8)))  float  v8f;
typedef __attribute__((ext_vector_type(2)))  __bf16 v2bf;

union FragU { v16bf v; uint4 u[2]; };

#define V8F_ZERO {0.f,0.f,0.f,0.f,0.f,0.f,0.f,0.f}

// Async global -> LDS copy of 16 bytes (gfx1250 GLOBAL_LOAD_ASYNC_TO_LDS_B128,
// tracked by ASYNCcnt). ldsoff is the 32-bit LDS byte address, gp the global VA.
__device__ __forceinline__ void async_b128(unsigned ldsoff, const void* gp) {
  asm volatile("global_load_async_to_lds_b128 %0, %1, off"
               :: "v"(ldsoff), "v"(gp) : "memory");
}
__device__ __forceinline__ void wait_async(int n) {
  if (n == 0)       asm volatile("s_wait_asynccnt 0" ::: "memory");
  else if (n == 6)  asm volatile("s_wait_asynccnt 6" ::: "memory");
  else              asm volatile("s_wait_asynccnt 0" ::: "memory");
}
__device__ __forceinline__ unsigned lds_off(const void* p) {
  return (unsigned)(unsigned long long)p;   // low 32 bits = LDS byte offset
}

// ---------------------------------------------------------------------------
// Elementwise: concat two f32 matrices [rows,C]|[rows,C] -> bf16 [rows,2C]
// ---------------------------------------------------------------------------
__global__ __launch_bounds__(256)
void concat_f32_to_bf16(const float* __restrict__ L, const float* __restrict__ R,
                        bf16* __restrict__ dst, long long rows, int C) {
  long long e = (long long)blockIdx.x * blockDim.x + threadIdx.x;
  long long total = rows * (long long)(2 * C);
  if (e >= total) return;
  long long r = e / (2 * C);
  int c = (int)(e - r * (2 * C));
  float v = (c < C) ? L[r * C + c] : R[r * C + (c - C)];
  dst[e] = (bf16)v;
}

// ---------------------------------------------------------------------------
// Elementwise: f32 -> bf16
// ---------------------------------------------------------------------------
__global__ __launch_bounds__(256)
void convert_f32_to_bf16(const float* __restrict__ src, bf16* __restrict__ dst,
                         long long n) {
  long long e = (long long)blockIdx.x * blockDim.x + threadIdx.x;
  if (e >= n) return;
  dst[e] = (bf16)src[e];
}

// ---------------------------------------------------------------------------
// bf16 WMMA GEMM: C[M,N] = A[M,K] * B[N,K]^T   (A,B bf16 row-major, C f32)
// Block tile 128x64x64, 256 threads = 8 waves (4x2), wave tile 32x32.
// Double-buffered LDS fed by GLOBAL_LOAD_ASYNC_TO_LDS_B128 (ASYNCcnt).
// Requires M%128==0, N%64==0, K%64==0.
// ---------------------------------------------------------------------------
#define G_BM 128
#define G_BN 64
#define G_BK 64
#define G_LD 72  // LDS row stride (bf16 elems), %8==0 keeps b128 alignment

__global__ __launch_bounds__(256)
void gemm_bf16_tn(const bf16* __restrict__ A, const bf16* __restrict__ B,
                  float* __restrict__ C, int M, int N, int K) {
  __shared__ bf16 As[2][G_BM * G_LD];   // 2 * 18432 B
  __shared__ bf16 Bs[2][G_BN * G_LD];   // 2 *  9216 B   (total 55296 B)

  const int tid  = threadIdx.x;
  const int wave = tid >> 5;
  const int lane = tid & 31;
  const int half = lane >> 4;
  const int ml   = lane & 15;
  const int wrow = wave >> 1;  // 0..3 -> 32-row stripe
  const int wcol = wave & 1;   // 0..1 -> 32-col stripe
  const int m0 = blockIdx.y * G_BM;
  const int n0 = blockIdx.x * G_BN;

  // Per-tile async staging: A = 1024 uint4 (4/thread), B = 512 uint4 (2/thread)
  auto issue_tile = [&](int buf, int k0) {
#pragma unroll
    for (int i = 0; i < 4; ++i) {
      int u = tid + i * 256;
      int r = u >> 3;            // 64 bf16 = 8 uint4 per row
      int c8 = (u & 7) * 8;
      async_b128(lds_off(&As[buf][r * G_LD + c8]),
                 &A[(size_t)(m0 + r) * K + k0 + c8]);
    }
#pragma unroll
    for (int i = 0; i < 2; ++i) {
      int u = tid + i * 256;
      int r = u >> 3;
      int c8 = (u & 7) * 8;
      async_b128(lds_off(&Bs[buf][r * G_LD + c8]),
                 &B[(size_t)(n0 + r) * K + k0 + c8]);
    }
  };

  v8f acc[2][2];
#pragma unroll
  for (int i = 0; i < 2; ++i)
#pragma unroll
    for (int j = 0; j < 2; ++j) acc[i][j] = (v8f)V8F_ZERO;

  const int nt = K / G_BK;
  issue_tile(0, 0);

  for (int t = 0; t < nt; ++t) {
    const int buf = t & 1;
    if (t + 1 < nt) {
      issue_tile(buf ^ 1, (t + 1) * G_BK);
      wait_async(6);   // the 6 ops of tile t are complete (in-order per wave)
    } else {
      wait_async(0);
    }
    __syncthreads();

#pragma unroll
    for (int ks = 0; ks < G_BK; ks += 32) {
      FragU a[2], b[2];
#pragma unroll
      for (int i = 0; i < 2; ++i) {
        int row = wrow * 32 + i * 16 + ml;
        a[i].u[0] = *(const uint4*)&As[buf][row * G_LD + ks + (half ? 8 : 0)];
        a[i].u[1] = *(const uint4*)&As[buf][row * G_LD + ks + (half ? 24 : 16)];
      }
#pragma unroll
      for (int j = 0; j < 2; ++j) {
        int row = wcol * 32 + j * 16 + ml;
        int kb = ks + (half ? 8 : 0);
        b[j].u[0] = *(const uint4*)&Bs[buf][row * G_LD + kb];
        b[j].u[1] = *(const uint4*)&Bs[buf][row * G_LD + kb + 16];
      }
#pragma unroll
      for (int i = 0; i < 2; ++i)
#pragma unroll
        for (int j = 0; j < 2; ++j)
          acc[i][j] = __builtin_amdgcn_wmma_f32_16x16x32_bf16(
              false, a[i].v, false, b[j].v, (short)0, acc[i][j], false, false);
    }
    __syncthreads();  // tile `buf` free for reuse (overwritten at t+2's issue)
  }

  // C fragment layout: element (m = r + 8*half, n = lane&15)
#pragma unroll
  for (int i = 0; i < 2; ++i) {
    int mbase = m0 + wrow * 32 + i * 16 + 8 * half;
#pragma unroll
    for (int j = 0; j < 2; ++j) {
      int ncol = n0 + wcol * 32 + j * 16 + ml;
#pragma unroll
      for (int r = 0; r < 8; ++r)
        C[(size_t)(mbase + r) * N + ncol] = acc[i][j][r];
    }
  }
}

// ---------------------------------------------------------------------------
// Fused per-head RMSNorm + RoPE (half-split), f32 in -> head-major bf16 out.
// One wave per (b, n, head); hd = 128, lane handles 4 contiguous elements.
// ---------------------------------------------------------------------------
__global__ __launch_bounds__(256)
void rmsnorm_rope(const float* __restrict__ src, const float* __restrict__ w,
                  const float* __restrict__ cosb, const float* __restrict__ sinb,
                  bf16* __restrict__ dst, int B, int N, int HH) {
  const int wave = threadIdx.x >> 5;
  const int lane = threadIdx.x & 31;
  long long idx = (long long)blockIdx.x * 8 + wave;
  long long total = (long long)B * N * HH;
  if (idx >= total) return;
  int hh = (int)(idx % HH);
  int n  = (int)((idx / HH) % N);
  int b  = (int)(idx / ((long long)HH * N));

  const float4 v =
      *(const float4*)&src[((size_t)(b * N + n) * HH + hh) * 128 + lane * 4];
  float ss = v.x * v.x + v.y * v.y + v.z * v.z + v.w * v.w;
#pragma unroll
  for (int off = 16; off > 0; off >>= 1) ss += __shfl_xor(ss, off, 32);
  float rms = rsqrtf(ss * (1.0f / 128.0f) + 1e-6f);

  const float4 wv = *(const float4*)&w[lane * 4];
  float t0 = v.x * rms * wv.x;
  float t1 = v.y * rms * wv.y;
  float t2 = v.z * rms * wv.z;
  float t3 = v.w * rms * wv.w;

  int j0 = 2 * lane, j1 = 2 * lane + 1;
  float c0 = cosb[n * 64 + j0], s0 = sinb[n * 64 + j0];
  float c1 = cosb[n * 64 + j1], s1 = sinb[n * 64 + j1];

  bf16* dp = dst + ((size_t)(b * HH + hh) * N + n) * 128;
  v2bf lo = {(bf16)(t0 * c0 - t1 * s0), (bf16)(t2 * c1 - t3 * s1)};
  v2bf hi = {(bf16)(t0 * s0 + t1 * c0), (bf16)(t2 * s1 + t3 * c1)};
  *(v2bf*)&dp[j0] = lo;
  *(v2bf*)&dp[64 + j0] = hi;
}

// ---------------------------------------------------------------------------
// V: f32 token-major [B*N, KVH*128] -> bf16 head-major [B, KVH, N, 128]
// ---------------------------------------------------------------------------
__global__ __launch_bounds__(256)
void v_relayout(const float* __restrict__ Vf, bf16* __restrict__ Vh,
                int B, int N, int KVHn) {
  long long e = (long long)blockIdx.x * blockDim.x + threadIdx.x;
  long long total = (long long)B * N * KVHn * 128;
  if (e >= total) return;
  int C = KVHn * 128;
  long long row = e / C;
  int c = (int)(e - row * C);
  int g = c >> 7, d = c & 127;
  int b = (int)(row / N), n = (int)(row % N);
  Vh[((size_t)(b * KVHn + g) * N + n) * 128 + d] = (bf16)Vf[e];
}

// ---------------------------------------------------------------------------
// Flash attention (non-causal, GQA). 128 threads = 4 waves; 64-query tile;
// each wave owns 16 query rows. S and P*V via v_wmma_f32_16x16x32_bf16.
// Q/K tiles staged with GLOBAL_LOAD_ASYNC_TO_LDS_B128; V transposed via VGPRs.
// ---------------------------------------------------------------------------
#define AT_BQ 64
#define AT_BK 64
#define HD    128
#define QLD   136  // LDS stride for Q/K tiles (bf16 elems)
#define VLD   72   // LDS stride for transposed V
#define PLD   72   // LDS stride for per-wave P scratch

__global__ __launch_bounds__(128)
void attn_flash(const bf16* __restrict__ Qh, const bf16* __restrict__ Kh,
                const bf16* __restrict__ Vh, bf16* __restrict__ Oc,
                int B, int N, int H, int KVH, int D) {
  __shared__ bf16 Qs[AT_BQ * QLD];
  __shared__ bf16 Ks[AT_BK * QLD];
  __shared__ bf16 Vt[HD * VLD];
  __shared__ bf16 Ps[4 * 16 * PLD];

  const int tid  = threadIdx.x;
  const int wave = tid >> 5;
  const int lane = tid & 31;
  const int half = lane >> 4;
  const int ml   = lane & 15;

  const int nqt = N / AT_BQ;
  const int qt = blockIdx.x % nqt;
  const int h  = (blockIdx.x / nqt) % H;
  const int b  = blockIdx.x / (nqt * H);
  const int g  = h / (H / KVH);

  const bf16* Qg = Qh + ((size_t)(b * H + h) * N + qt * AT_BQ) * HD;
  const bf16* Kg = Kh + ((size_t)(b * KVH + g) * N) * HD;
  const bf16* Vg = Vh + ((size_t)(b * KVH + g) * N) * HD;

  // Q tile: rows 0..31 only, 32x128 bf16 = 512 uint4, async direct to LDS (4 per thread)
#pragma unroll
  for (int i = 0; i < 4; ++i) {
    int u = tid + i * 128;
    int r = u >> 4, c8 = (u & 15) * 8;
    async_b128(lds_off(&Qs[r * QLD + c8]), &Qg[(size_t)r * HD + c8]);
  }

  float mrow[8], lrow[8];
  v8f o[8];
#pragma unroll
  for (int r = 0; r < 8; ++r) { mrow[r] = -1e30f; lrow[r] = 0.f; }
#pragma unroll
  for (int nh = 0; nh < 8; ++nh) o[nh] = (v8f)V8F_ZERO;

  const float scale = 0.08838834764831845f;  // 128^-0.5

  for (int kt = 0; kt < N; kt += AT_BK) {
    // K tile rows kt..kt+31 only (32x128), async direct to LDS
#pragma unroll
    for (int i = 0; i < 4; ++i) {
      int u = tid + i * 128;
      int r = u >> 4, c8 = (u & 15) * 8;
      async_b128(lds_off(&Ks[r * QLD + c8]),
                 &Kg[(size_t)(kt + r) * HD + c8]);
    }
    // V tile 64x128, stored transposed as Vt[d][kv] (needs VGPR bounce)
#pragma unroll
    for (int i = 0; i < 8; ++i) {
      int u = tid + i * 128;
      int kv = u >> 4, d0 = (u & 15) * 8;
      union { uint4 q; bf16 hv[8]; } vv;
      vv.q = *(const uint4*)&Vg[(size_t)(kt + kv) * HD + d0];
#pragma unroll
      for (int j = 0; j < 8; ++j) Vt[(d0 + j) * VLD + kv] = vv.hv[j];
    }
    wait_async(0);      // Q (first iter) + K tile resident in LDS
    __syncthreads();

    // ---- S = Q K^T : wave's 16 rows x 64 kv, K-dim = 128 (4 wmma steps)
    v8f s[4];
#pragma unroll
    for (int sub = 0; sub < 4; ++sub) s[sub] = (v8f)V8F_ZERO;
#pragma unroll
    for (int kk = 0; kk < 4; ++kk) {
      FragU aq;
      int qrow = wave * 16 + ml;
      aq.u[0] = *(const uint4*)&Qs[qrow * QLD + kk * 32 + (half ? 8 : 0)];
      aq.u[1] = *(const uint4*)&Qs[qrow * QLD + kk * 32 + (half ? 24 : 16)];
#pragma unroll
      for (int sub = 0; sub < 4; ++sub) {
        FragU bk;
        int krow = sub * 16 + ml;
        int kb = kk * 32 + (half ? 8 : 0);
        bk.u[0] = *(const uint4*)&Ks[krow * QLD + kb];
        bk.u[1] = *(const uint4*)&Ks[krow * QLD + kb + 16];
        s[sub] = __builtin_amdgcn_wmma_f32_16x16x32_bf16(
            false, aq.v, false, bk.v, (short)0, s[sub], false, false);
      }
    }

    // ---- online softmax on C-fragment layout (row = r + 8*half)
#pragma unroll
    for (int r = 0; r < 8; ++r) {
      float lm = -1e30f;
#pragma unroll
      for (int sub = 0; sub < 4; ++sub) {
        s[sub][r] *= scale;
        lm = fmaxf(lm, s[sub][r]);
      }
#pragma unroll
      for (int off = 8; off > 0; off >>= 1)
        lm = fmaxf(lm, __shfl_xor(lm, off, 32));  // stays in 16-lane N group
      float nm = fmaxf(mrow[r], lm);
      float alpha = __expf(mrow[r] - nm);
      mrow[r] = nm;
      float rs = 0.f;
#pragma unroll
      for (int sub = 0; sub < 4; ++sub) {
        float p = __expf(s[sub][r] - nm);
        s[sub][r] = p;
        rs += p;
      }
#pragma unroll
      for (int off = 8; off > 0; off >>= 1) rs += __shfl_xor(rs, off, 32);
      lrow[r] = lrow[r] * alpha + rs;
#pragma unroll
      for (int nh = 0; nh < 8; ++nh) o[nh][r] *= alpha;
    }

    // ---- P: C-layout -> LDS -> A-layout (wave-private region)
    bf16* Pw = &Ps[wave * 16 * PLD];
#pragma unroll
    for (int sub = 0; sub < 4; ++sub)
#pragma unroll
      for (int r = 0; r < 8; ++r)
        Pw[(r + 8 * half) * PLD + sub * 16 + ml] = (bf16)s[sub][r];

    // ---- O += P V : M=16, K=64 (2 wmma steps), N=128 (8 tiles)
#pragma unroll
    for (int k2 = 0; k2 < 2; ++k2) {
      FragU ap;
      ap.u[0] = *(const uint4*)&Pw[ml * PLD + k2 * 32 + (half ? 8 : 0)];
      ap.u[1] = *(const uint4*)&Pw[ml * PLD + k2 * 32 + (half ? 24 : 16)];
#pragma unroll
      for (int nh = 0; nh < 8; ++nh) {
        FragU bv;
        int vrow = nh * 16 + ml;
        int kb = k2 * 32 + (half ? 8 : 0);
        bv.u[0] = *(const uint4*)&Vt[vrow * VLD + kb];
        bv.u[1] = *(const uint4*)&Vt[vrow * VLD + kb + 16];
        o[nh] = __builtin_amdgcn_wmma_f32_16x16x32_bf16(
            false, ap.v, false, bv.v, (short)0, o[nh], false, false);
      }
    }
    __syncthreads();
  }

  // ---- epilogue: O / l, write token-major bf16 [B*N, D]
#pragma unroll
  for (int nh = 0; nh < 8; ++nh) {
    int col = h * HD + nh * 16 + ml;
#pragma unroll
    for (int r = 0; r < 8; ++r) {
      int n_tok = qt * AT_BQ + wave * 16 + r + 8 * half;
      Oc[(size_t)(b * N + n_tok) * D + col] = (bf16)(o[nh][r] / lrow[r]);
    }
  }
}

// ---------------------------------------------------------------------------
// Host-side orchestration
// ---------------------------------------------------------------------------
static inline size_t align256(size_t x) { return (x + 255) & ~(size_t)255; }
static inline int ceil_div_ll(long long a, int b) { return (int)((a + b - 1) / b); }

extern "C" void kernel_launch(void* const* d_in, const int* in_sizes, int n_in,
                              void* d_out, int out_size, void* d_ws, size_t ws_size,
                              hipStream_t stream) {
  const float* x        = (const float*)d_in[0];
  const float* mu_prev  = (const float*)d_in[1];
  const float* cosb     = (const float*)d_in[2];
  const float* sinb     = (const float*)d_in[3];
  const float* wq       = (const float*)d_in[4];
  const float* wk       = (const float*)d_in[5];
  const float* wv       = (const float*)d_in[6];
  const float* mu_q_w   = (const float*)d_in[7];
  const float* mu_k_w   = (const float*)d_in[8];
  const float* mu_v_w   = (const float*)d_in[9];
  const float* wo       = (const float*)d_in[10];
  const float* q_norm_w = (const float*)d_in[11];
  const float* k_norm_w = (const float*)d_in[12];

  const int D = 2048, H = 16, KVH = 4, HDc = 128;
  const int KV = KVH * HDc;                 // 512
  const int N = in_sizes[2] / (HDc / 2);    // cos is (N, 64)
  const int B = in_sizes[0] / (N * D);
  const int M = B * N;                      // 4096 tokens
  const int K2 = 2 * D;                     // 4096 concat dim

  // workspace partition
  char* ws = (char*)d_ws;
  size_t off = 0;
  auto take = [&](size_t bytes) { void* p = ws + off; off = align256(off + bytes); return p; };
  bf16*  Xc  = (bf16*)take((size_t)M * K2 * 2);
  bf16*  WqC = (bf16*)take((size_t)D * K2 * 2);
  bf16*  WkC = (bf16*)take((size_t)KV * K2 * 2);
  bf16*  WvC = (bf16*)take((size_t)KV * K2 * 2);
  bf16*  WoC = (bf16*)take((size_t)D * D * 2);
  float* Qf  = (float*)take((size_t)M * D * 4);
  float* Kf  = (float*)take((size_t)M * KV * 4);
  float* Vf  = (float*)take((size_t)M * KV * 4);
  bf16*  Qhd = (bf16*)take((size_t)B * H * N * HDc * 2);
  bf16*  Khd = (bf16*)take((size_t)B * KVH * N * HDc * 2);
  bf16*  Vhd = (bf16*)take((size_t)B * KVH * N * HDc * 2);
  bf16*  Oc  = (bf16*)take((size_t)M * D * 2);
  (void)ws_size; (void)n_in; (void)out_size;

  // 1) bf16 staging
  concat_f32_to_bf16<<<ceil_div_ll((long long)M * K2, 256), 256, 0, stream>>>(
      x, mu_prev, Xc, M, D);
  concat_f32_to_bf16<<<ceil_div_ll((long long)D * K2, 256), 256, 0, stream>>>(
      wq, mu_q_w, WqC, D, D);
  concat_f32_to_bf16<<<ceil_div_ll((long long)KV * K2, 256), 256, 0, stream>>>(
      wk, mu_k_w, WkC, KV, D);
  concat_f32_to_bf16<<<ceil_div_ll((long long)KV * K2, 256), 256, 0, stream>>>(
      wv, mu_v_w, WvC, KV, D);
  convert_f32_to_bf16<<<ceil_div_ll((long long)D * D, 256), 256, 0, stream>>>(
      wo, WoC, (long long)D * D);

  // 2) QKV projections (WMMA GEMMs)
  gemm_bf16_tn<<<dim3(D / G_BN, M / G_BM), 256, 0, stream>>>(Xc, WqC, Qf, M, D, K2);
  gemm_bf16_tn<<<dim3(KV / G_BN, M / G_BM), 256, 0, stream>>>(Xc, WkC, Kf, M, KV, K2);
  gemm_bf16_tn<<<dim3(KV / G_BN, M / G_BM), 256, 0, stream>>>(Xc, WvC, Vf, M, KV, K2);

  // 3) RMSNorm + RoPE -> head-major bf16; V relayout
  rmsnorm_rope<<<ceil_div_ll((long long)B * N * H, 8), 256, 0, stream>>>(
      Qf, q_norm_w, cosb, sinb, Qhd, B, N, H);
  rmsnorm_rope<<<ceil_div_ll((long long)B * N * KVH, 8), 256, 0, stream>>>(
      Kf, k_norm_w, cosb, sinb, Khd, B, N, KVH);
  v_relayout<<<ceil_div_ll((long long)M * KV, 256), 256, 0, stream>>>(
      Vf, Vhd, B, N, KVH);

  // 4) flash attention (WMMA), writes token-major bf16 O
  attn_flash<<<B * H * (N / AT_BQ), 128, 0, stream>>>(
      Qhd, Khd, Vhd, Oc, B, N, H, KVH, D);

  // 5) output projection -> f32 d_out
  gemm_bf16_tn<<<dim3(D / G_BN, M / G_BM), 256, 0, stream>>>(
      Oc, WoC, (float*)d_out, M, D, D);
}
